// LinkPredictor_50955492000035
// MI455X (gfx1250) — matrix operand order repaired and checked by reading, not compiled, not run on hardware
//
#include <hip/hip_runtime.h>
#include <hip/hip_bf16.h>

// ---------------------------------------------------------------------------
// LinkPredictor forward on gfx1250 (MI455X).
// All matmuls on v_wmma_f32_16x16x32_bf16 (f32 accumulate). GEMM B-tiles are
// staged into LDS with global_load_async_to_lds_b128 (ASYNCcnt), double
// buffered; A fragments are register double-buffered; B fragments are
// preloaded in groups of 4 so WMMAs issue back-to-back.
// ---------------------------------------------------------------------------

typedef __bf16 bf16_t;
typedef __attribute__((ext_vector_type(16))) __bf16 v16bf;
typedef __attribute__((ext_vector_type(8)))  __bf16 v8bf;
typedef __attribute__((ext_vector_type(8)))  float  v8f;

#define WMMA_BF16(a, b, c) \
  __builtin_amdgcn_wmma_f32_16x16x32_bf16(false, (a), false, (b), (short)0, (c), false, false)

// Constants from the reference
#define BSZ   512
#define NN    64
#define DIM   512
#define HEADS 8
#define HD    64
#define EDGE  4096
#define CDIM  128
#define CELLD 64
#define DD    1152   // CDIM + 2*DIM

// ---------------------------------------------------------------------------
// Fragment loaders
// A-matrix 16x32 bf16 (ISA 7.12.2): lane holds row M = lane%16, half = lane/16
//   element e<8 : K = k0 + half*8 + e ; e>=8: K = k0 + 16 + half*8 + (e-8)
// ---------------------------------------------------------------------------
__device__ inline v16bf load_frag_a(const bf16_t* __restrict__ base, int ld,
                                    int r, int k0, int lane) {
  int half = lane >> 4;
  const bf16_t* p = base + (size_t)r * ld + k0 + half * 8;
  union { v16bf v; v8bf h[2]; } u;
  u.h[0] = *(const v8bf*)(p);
  u.h[1] = *(const v8bf*)(p + 16);
  return u.v;
}

// B-matrix 32x16 bf16: lane holds col N = lane%16, half = lane/16
//   element e<8 : K = k0 + half*8 + e ; e>=8: K = k0 + 16 + half*8 + (e-8)
__device__ inline v16bf load_frag_b(const bf16_t* __restrict__ base, int ld,
                                    int n, int k0, int lane) {
  int half = lane >> 4;
  const bf16_t* p = base + (size_t)n * ld + k0 + half * 8;
  union { v16bf v; v8bf h[2]; } u;
  u.h[0] = *(const v8bf*)(p);
  u.h[1] = *(const v8bf*)(p + 16);
  return u.v;
}

__device__ inline float block_reduce_sum(float v, float* red, int tid, int nthr) {
  red[tid] = v;
  __syncthreads();
  for (int off = nthr >> 1; off > 0; off >>= 1) {
    if (tid < off) red[tid] += red[tid + off];
    __syncthreads();
  }
  float r = red[0];
  __syncthreads();
  return r;
}

// ---------------------------------------------------------------------------
// bf16 WMMA GEMM with async-LDS staged, double-buffered B tiles.
// out[M,N] = act(A[M,K] @ W[N,K]^T + bias) (+ residual)
// block = 256 threads (8 waves); tile BM=128 x BN=64; K stage = 64.
// ---------------------------------------------------------------------------
#define GBM 128
#define GBN 64
#define BLD 72   // LDS row pitch (elements) for the B tile

__global__ __launch_bounds__(256) void gemm_bf16_kernel(
    const bf16_t* __restrict__ A, const bf16_t* __restrict__ W,
    const float* __restrict__ bias, const float* __restrict__ residual,
    void* __restrict__ out, int M, int N, int K, int relu, int out_bf16) {
  __shared__ __align__(16) bf16_t Bs[2][GBN][BLD];

  int lane = threadIdx.x & 31;
  int wave = threadIdx.x >> 5;
  int row0 = blockIdx.x * GBM + wave * 16;
  int col0 = blockIdx.y * GBN;
  int r = row0 + (lane & 15);

  unsigned int lds_base = (unsigned int)(uintptr_t)&Bs[0][0][0];

  // One 64-col x 64-K B stage = 8KB = 512 x 16B chunks; 256 threads -> 2 each.
  // Each wave issues 2 async instructions (ASYNCcnt += 2 per stage).
  auto issue_stage = [&](int buf, int kt) {
#pragma unroll
    for (int i = 0; i < 2; ++i) {
      int q = i * 256 + (int)threadIdx.x;          // 0..511
      int n = q >> 3;                              // B row (column of W^T)
      int kc = q & 7;                              // 16B chunk within the row
      unsigned int lds =
          lds_base + (unsigned int)(((buf * GBN + n) * BLD + kc * 8) * 2);
      unsigned long long ga =
          (unsigned long long)(const void*)(W + (size_t)(col0 + n) * K + kt + kc * 8);
      asm volatile("global_load_async_to_lds_b128 %0, %1, off"
                   :: "v"(lds), "v"(ga) : "memory");
    }
  };

  v8f acc[4] = {};

  // Preload all 4 B fragments of a 32-K step, then 4 back-to-back WMMAs.
  auto compute_stage = [&](int buf, v16bf a0, v16bf a1) {
    const bf16_t* bsrc = &Bs[buf][0][0];
    v16bf b0, b1, b2, b3;
    int nn = lane & 15;
    b0 = load_frag_b(bsrc, BLD, nn,      0, lane);
    b1 = load_frag_b(bsrc, BLD, nn + 16, 0, lane);
    b2 = load_frag_b(bsrc, BLD, nn + 32, 0, lane);
    b3 = load_frag_b(bsrc, BLD, nn + 48, 0, lane);
    acc[0] = WMMA_BF16(a0, b0, acc[0]);
    acc[1] = WMMA_BF16(a0, b1, acc[1]);
    acc[2] = WMMA_BF16(a0, b2, acc[2]);
    acc[3] = WMMA_BF16(a0, b3, acc[3]);
    b0 = load_frag_b(bsrc, BLD, nn,      32, lane);
    b1 = load_frag_b(bsrc, BLD, nn + 16, 32, lane);
    b2 = load_frag_b(bsrc, BLD, nn + 32, 32, lane);
    b3 = load_frag_b(bsrc, BLD, nn + 48, 32, lane);
    acc[0] = WMMA_BF16(a1, b0, acc[0]);
    acc[1] = WMMA_BF16(a1, b1, acc[1]);
    acc[2] = WMMA_BF16(a1, b2, acc[2]);
    acc[3] = WMMA_BF16(a1, b3, acc[3]);
  };

  issue_stage(0, 0);
  v16bf a0 = load_frag_a(A, K, r, 0, lane);
  v16bf a1 = load_frag_a(A, K, r, 32, lane);

  int buf = 0;
  int kt = 0;
  // Steady state: next stage always exists -> unconditional wait 0x2.
  for (; kt + 64 < K; kt += 64, buf ^= 1) {
    issue_stage(buf ^ 1, kt + 64);
    v16bf na0 = load_frag_a(A, K, r, kt + 64, lane);
    v16bf na1 = load_frag_a(A, K, r, kt + 96, lane);
    asm volatile("s_wait_asynccnt 0x2" ::: "memory");
    __syncthreads();
    compute_stage(buf, a0, a1);
    __syncthreads();  // all waves done reading buf before it is refilled
    a0 = na0;
    a1 = na1;
  }
  // Epilogue: last stage, drain ASYNCcnt fully.
  asm volatile("s_wait_asynccnt 0x0" ::: "memory");
  __syncthreads();
  compute_stage(buf, a0, a1);

  int rhi = (lane >> 4) * 8;
  int cl = lane & 15;
#pragma unroll
  for (int cc = 0; cc < 4; ++cc) {
    int col = col0 + cc * 16 + cl;
    float bv = bias ? bias[col] : 0.0f;
#pragma unroll
    for (int i = 0; i < 8; ++i) {
      int row = row0 + rhi + i;
      float v = acc[cc][i] + bv;
      if (relu) v = fmaxf(v, 0.0f);
      if (residual) v += residual[(size_t)row * N + col];
      if (out_bf16) ((bf16_t*)out)[(size_t)row * N + col] = (bf16_t)v;
      else          ((float*)out)[(size_t)row * N + col] = v;
    }
  }
}

// ---------------------------------------------------------------------------
// LayerNorm over last dim C, optional activation (0 none, 1 relu, 2 tanh),
// output f32 or bf16. One block (256 thr) per row.
// ---------------------------------------------------------------------------
__global__ __launch_bounds__(256) void ln_act_kernel(
    const float* __restrict__ x, const float* __restrict__ g,
    const float* __restrict__ bb, void* __restrict__ out,
    int C, int act, int out_bf16) {
  __shared__ float red[256];
  int row = blockIdx.x;
  const float* xr = x + (size_t)row * C;
  float s = 0.f, ss = 0.f;
  for (int j = threadIdx.x; j < C; j += 256) {
    float v = xr[j];
    s += v; ss += v * v;
  }
  float mean = block_reduce_sum(s, red, threadIdx.x, 256) / (float)C;
  float var  = block_reduce_sum(ss, red, threadIdx.x, 256) / (float)C - mean * mean;
  float rstd = rsqrtf(var + 1e-5f);
  for (int j = threadIdx.x; j < C; j += 256) {
    float v = (xr[j] - mean) * rstd * g[j] + bb[j];
    if (act == 1) v = fmaxf(v, 0.0f);
    else if (act == 2) v = tanhf(v);
    if (out_bf16) ((bf16_t*)out)[(size_t)row * C + j] = (bf16_t)v;
    else          ((float*)out)[(size_t)row * C + j] = v;
  }
}

// H[b,n,:] = vidx_emb[vidx] + pos_emb[pos] + deg_emb[deg]
__global__ void embed_kernel(const int* __restrict__ vidx,
                             const int* __restrict__ pos,
                             const int* __restrict__ deg,
                             const float* __restrict__ ve,
                             const float* __restrict__ pe,
                             const float* __restrict__ de,
                             float* __restrict__ H) {
  int idx = blockIdx.x * 256 + threadIdx.x;
  int bn = idx >> 9;
  int d = idx & 511;
  H[idx] = ve[(size_t)vidx[bn] * DIM + d] + pe[(size_t)pos[bn] * DIM + d] +
           de[(size_t)deg[bn] * DIM + d];
}

__global__ void cast_bf16_kernel(const float* __restrict__ x,
                                 bf16_t* __restrict__ y, int n) {
  int i = blockIdx.x * 256 + threadIdx.x;
  if (i < n) y[i] = (bf16_t)x[i];
}

// ---------------------------------------------------------------------------
// Fused attention for one (head, batch): S = QK^T + bias, mask, *1/8,
// softmax, P@V -> attn_out (bf16, [b,n,dim] layout). 128 threads.
// V tile is cooperatively transposed into LDS so P@V B-fragments are two
// ds_load_b128 instead of 16 scalar global loads.
// ---------------------------------------------------------------------------
__global__ __launch_bounds__(128) void attn_kernel(
    const bf16_t* __restrict__ qbf, const bf16_t* __restrict__ kbf,
    const bf16_t* __restrict__ vbf, const float* __restrict__ bias,
    const int* __restrict__ mask, bf16_t* __restrict__ attn_out) {
  __shared__ float sc[NN][72];
  __shared__ __align__(16) bf16_t pb[NN][72];
  __shared__ __align__(16) bf16_t vt[HD][72];   // vt[d][key] = V[key][d]

  int h = blockIdx.x & 7;
  int b = blockIdx.x >> 3;
  int lane = threadIdx.x & 31;
  int wave = threadIdx.x >> 5;
  int row0 = wave * 16;
  int r = row0 + (lane & 15);

  const bf16_t* qb = qbf + (size_t)b * NN * DIM + h * HD;
  const bf16_t* kb = kbf + (size_t)b * NN * DIM + h * HD;
  const bf16_t* vb = vbf + (size_t)b * NN * DIM + h * HD;

  // --- cooperative V transpose into LDS (coalesced 16B global reads) ---
  // 64 keys x 8 chunks(16B) = 512 chunks; 128 threads -> 4 each.
#pragma unroll
  for (int i = 0; i < 4; ++i) {
    int q = i * 128 + (int)threadIdx.x;  // 0..511
    int key = q >> 3;
    int dc = q & 7;
    v8bf vv = *(const v8bf*)(vb + (size_t)key * DIM + dc * 8);
#pragma unroll
    for (int e = 0; e < 8; ++e) vt[dc * 8 + e][key] = vv[e];
  }

  // --- S = Q K^T ---
  v8f acc[4] = {};
  for (int k0 = 0; k0 < HD; k0 += 32) {
    v16bf a = load_frag_a(qb, DIM, r, k0, lane);
    v16bf b0 = load_frag_b(kb, DIM, (lane & 15),      k0, lane);
    v16bf b1 = load_frag_b(kb, DIM, (lane & 15) + 16, k0, lane);
    v16bf b2 = load_frag_b(kb, DIM, (lane & 15) + 32, k0, lane);
    v16bf b3 = load_frag_b(kb, DIM, (lane & 15) + 48, k0, lane);
    acc[0] = WMMA_BF16(a, b0, acc[0]);
    acc[1] = WMMA_BF16(a, b1, acc[1]);
    acc[2] = WMMA_BF16(a, b2, acc[2]);
    acc[3] = WMMA_BF16(a, b3, acc[3]);
  }
  const float* brow = bias + (size_t)b * NN * NN;
  const int* mrow = mask + (size_t)b * NN * NN;
  int rhi = (lane >> 4) * 8;
  int cl = lane & 15;
  const float scale = 0.125f;  // 1/sqrt(HD)
#pragma unroll
  for (int cc = 0; cc < 4; ++cc) {
    int col = cc * 16 + cl;
#pragma unroll
    for (int i = 0; i < 8; ++i) {
      int row = row0 + rhi + i;
      float v = acc[cc][i] + brow[row * NN + col];
      if (mrow[row * NN + col] == 0) v = -1e20f;
      sc[row][col] = v * scale;
    }
  }
  __syncthreads();

  // --- softmax rows (threads 0..63, one row each) ---
  if (threadIdx.x < NN) {
    int row = threadIdx.x;
    float m = -3.4e38f;
    for (int j = 0; j < NN; ++j) m = fmaxf(m, sc[row][j]);
    float sum = 0.f;
    for (int j = 0; j < NN; ++j) sum += __expf(sc[row][j] - m);
    float inv = 1.0f / sum;
    for (int j = 0; j < NN; ++j)
      pb[row][j] = (bf16_t)(__expf(sc[row][j] - m) * inv);
  }
  __syncthreads();

  // --- attn = P @ V  (both operands from LDS) ---
  v8f oacc[4] = {};
  for (int k0 = 0; k0 < NN; k0 += 32) {
    v16bf a = load_frag_a(&pb[0][0], 72, r, k0, lane);
    v16bf b0 = load_frag_b(&vt[0][0], 72, (lane & 15),      k0, lane);
    v16bf b1 = load_frag_b(&vt[0][0], 72, (lane & 15) + 16, k0, lane);
    v16bf b2 = load_frag_b(&vt[0][0], 72, (lane & 15) + 32, k0, lane);
    v16bf b3 = load_frag_b(&vt[0][0], 72, (lane & 15) + 48, k0, lane);
    oacc[0] = WMMA_BF16(a, b0, oacc[0]);
    oacc[1] = WMMA_BF16(a, b1, oacc[1]);
    oacc[2] = WMMA_BF16(a, b2, oacc[2]);
    oacc[3] = WMMA_BF16(a, b3, oacc[3]);
  }
  bf16_t* ob = attn_out + (size_t)b * NN * DIM + h * HD;
#pragma unroll
  for (int cc = 0; cc < 4; ++cc) {
    int col = cc * 16 + cl;
#pragma unroll
    for (int i = 0; i < 8; ++i) {
      int row = row0 + rhi + i;
      ob[(size_t)row * DIM + col] = (bf16_t)oacc[cc][i];
    }
  }
}

// cell_feat gather: cf[b,:] = cell_feat[c0, c1, :]
__global__ void cell_gather_kernel(const int* __restrict__ cidx,
                                   const float* __restrict__ cell_feat,
                                   float* __restrict__ cf) {
  int b = blockIdx.x;
  int j = threadIdx.x;  // 64 threads
  int c0 = cidx[b * 2 + 0];
  int c1 = cidx[b * 2 + 1];
  cf[(size_t)b * CELLD + j] = cell_feat[((size_t)c0 * 200 + c1) * CELLD + j];
}

// decoder concat: xcat = [g(f32,already LN), LN(H[:,0]), LN(H[:,1])] as bf16
__global__ __launch_bounds__(256) void dec_concat_kernel(
    const float* __restrict__ g, const float* __restrict__ H,
    const float* __restrict__ lnfg, const float* __restrict__ lnfb,
    bf16_t* __restrict__ xcat) {
  __shared__ float red[256];
  int b = blockIdx.x;
  if (threadIdx.x < CDIM)
    xcat[(size_t)b * DD + threadIdx.x] = (bf16_t)g[(size_t)b * CDIM + threadIdx.x];
  for (int sel = 0; sel < 2; ++sel) {
    const float* xr = H + ((size_t)b * NN + sel) * DIM;
    float s = 0.f, ss = 0.f;
    for (int j = threadIdx.x; j < DIM; j += 256) {
      float v = xr[j];
      s += v; ss += v * v;
    }
    float mean = block_reduce_sum(s, red, threadIdx.x, 256) / (float)DIM;
    float var  = block_reduce_sum(ss, red, threadIdx.x, 256) / (float)DIM - mean * mean;
    float rstd = rsqrtf(var + 1e-5f);
    for (int j = threadIdx.x; j < DIM; j += 256)
      xcat[(size_t)b * DD + CDIM + sel * DIM + j] =
          (bf16_t)((xr[j] - mean) * rstd * lnfg[j] + lnfb[j]);
    __syncthreads();
  }
}

// y[b] = sigmoid(h1[b,:] . W2 + b2)
__global__ __launch_bounds__(128) void dec_final_kernel(
    const bf16_t* __restrict__ h1, const float* __restrict__ W2,
    const float* __restrict__ b2, float* __restrict__ y) {
  __shared__ float red[128];
  int b = blockIdx.x;
  float s = 0.f;
  for (int j = threadIdx.x; j < DD; j += 128)
    s += (float)h1[(size_t)b * DD + j] * W2[j];
  float dot = block_reduce_sum(s, red, threadIdx.x, 128);
  if (threadIdx.x == 0) y[b] = 1.0f / (1.0f + __expf(-(dot + b2[0])));
}

// ---------------------------------------------------------------------------
// Host orchestration
// ---------------------------------------------------------------------------
static inline void cast_w(const float* src, bf16_t* dst, int n, hipStream_t s) {
  cast_bf16_kernel<<<(n + 255) / 256, 256, 0, s>>>(src, dst, n);
}

extern "C" void kernel_launch(void* const* d_in, const int* in_sizes, int n_in,
                              void* d_out, int out_size, void* d_ws, size_t ws_size,
                              hipStream_t stream) {
  (void)in_sizes; (void)n_in; (void)out_size; (void)ws_size;

  // ---- inputs (setup_inputs dict order, params flattened in insertion order)
  const int* vidx = (const int*)d_in[0];
  const int* pos  = (const int*)d_in[1];
  const int* deg  = (const int*)d_in[2];
  const float* dist = (const float*)d_in[3];
  const int* mask = (const int*)d_in[4];
  const int* cidx = (const int*)d_in[5];
  int p = 6;
  const float* vemb = (const float*)d_in[p++];
  const float* pemb = (const float*)d_in[p++];
  const float* demb = (const float*)d_in[p++];
  const float* e_ln0g = (const float*)d_in[p++];
  const float* e_ln0b = (const float*)d_in[p++];
  const float* e_W1 = (const float*)d_in[p++];
  const float* e_b1 = (const float*)d_in[p++];
  const float* e_ln1g = (const float*)d_in[p++];
  const float* e_ln1b = (const float*)d_in[p++];
  const float* e_W2 = (const float*)d_in[p++];
  const float* e_b2 = (const float*)d_in[p++];
  const float* e_ln2g = (const float*)d_in[p++];
  const float* e_ln2b = (const float*)d_in[p++];
  struct Layer {
    const float *n1g, *n1b, *Wq, *bq, *Wk, *bk, *Wv, *bv, *Wo, *bo;
    const float *n2g, *n2b, *Wf1, *bf1, *Wf2, *bf2;
  } L[4];
  for (int l = 0; l < 4; ++l) {
    L[l].n1g = (const float*)d_in[p++]; L[l].n1b = (const float*)d_in[p++];
    L[l].Wq = (const float*)d_in[p++];  L[l].bq = (const float*)d_in[p++];
    L[l].Wk = (const float*)d_in[p++];  L[l].bk = (const float*)d_in[p++];
    L[l].Wv = (const float*)d_in[p++];  L[l].bv = (const float*)d_in[p++];
    L[l].Wo = (const float*)d_in[p++];  L[l].bo = (const float*)d_in[p++];
    L[l].n2g = (const float*)d_in[p++]; L[l].n2b = (const float*)d_in[p++];
    L[l].Wf1 = (const float*)d_in[p++]; L[l].bf1 = (const float*)d_in[p++];
    L[l].Wf2 = (const float*)d_in[p++]; L[l].bf2 = (const float*)d_in[p++];
  }
  const float* lnfg = (const float*)d_in[p++];
  const float* lnfb = (const float*)d_in[p++];
  const float* cell_feat = (const float*)d_in[p++];
  const float* cW1 = (const float*)d_in[p++];
  const float* cb1 = (const float*)d_in[p++];
  const float* cln1g = (const float*)d_in[p++];
  const float* cln1b = (const float*)d_in[p++];
  const float* cW2 = (const float*)d_in[p++];
  const float* cb2 = (const float*)d_in[p++];
  const float* cln2g = (const float*)d_in[p++];
  const float* cln2b = (const float*)d_in[p++];
  const float* dW1 = (const float*)d_in[p++];
  const float* db1 = (const float*)d_in[p++];
  const float* dW2 = (const float*)d_in[p++];
  const float* db2 = (const float*)d_in[p++];

  // ---- workspace carve-up
  char* w = (char*)d_ws;
  size_t off = 0;
  auto alloc = [&](size_t bytes) -> void* {
    void* r = (void*)(w + off);
    off = (off + bytes + 255) & ~(size_t)255;
    return r;
  };
  const size_t TOK = (size_t)BSZ * NN;  // 32768
  float*  H      = (float*)alloc(TOK * DIM * 4);
  bf16_t* xn_bf  = (bf16_t*)alloc(TOK * DIM * 2);
  bf16_t* q_bf   = (bf16_t*)alloc(TOK * DIM * 2);
  bf16_t* k_bf   = (bf16_t*)alloc(TOK * DIM * 2);
  bf16_t* v_bf   = (bf16_t*)alloc(TOK * DIM * 2);
  bf16_t* at_bf  = (bf16_t*)alloc(TOK * DIM * 2);
  bf16_t* f1_bf  = (bf16_t*)alloc(TOK * DIM * 2);
  float*  biasB  = (float*)alloc((size_t)BSZ * EDGE * 4);
  float*  z_f32  = (float*)alloc((size_t)BSZ * EDGE * 4);
  bf16_t* z_bf   = (bf16_t*)alloc((size_t)BSZ * EDGE * 2);
  float*  cf_f32 = (float*)alloc((size_t)BSZ * CELLD * 4);
  bf16_t* cf_bf  = (bf16_t*)alloc((size_t)BSZ * CELLD * 2);
  float*  g1_f32 = (float*)alloc((size_t)BSZ * CELLD * 4);
  bf16_t* g1_bf  = (bf16_t*)alloc((size_t)BSZ * CELLD * 2);
  float*  g2_f32 = (float*)alloc((size_t)BSZ * CDIM * 4);
  float*  gln    = (float*)alloc((size_t)BSZ * CDIM * 4);
  bf16_t* xcat   = (bf16_t*)alloc((size_t)BSZ * DD * 2);
  bf16_t* h1_bf  = (bf16_t*)alloc((size_t)BSZ * DD * 2);
  bf16_t* eW1_bf = (bf16_t*)alloc((size_t)EDGE * EDGE * 2);
  bf16_t* eW2_bf = (bf16_t*)alloc((size_t)EDGE * EDGE * 2);
  bf16_t* lw_bf[4][6];
  for (int l = 0; l < 4; ++l)
    for (int m = 0; m < 6; ++m)
      lw_bf[l][m] = (bf16_t*)alloc((size_t)DIM * DIM * 2);
  bf16_t* cW1_bf = (bf16_t*)alloc((size_t)CELLD * CELLD * 2);
  bf16_t* cW2_bf = (bf16_t*)alloc((size_t)CDIM * CELLD * 2);
  bf16_t* dW1_bf = (bf16_t*)alloc((size_t)DD * DD * 2);

  // ---- weight casts (f32 -> bf16)
  cast_w(e_W1, eW1_bf, EDGE * EDGE, stream);
  cast_w(e_W2, eW2_bf, EDGE * EDGE, stream);
  for (int l = 0; l < 4; ++l) {
    const float* srcs[6] = {L[l].Wq, L[l].Wk, L[l].Wv, L[l].Wo, L[l].Wf1, L[l].Wf2};
    for (int m = 0; m < 6; ++m) cast_w(srcs[m], lw_bf[l][m], DIM * DIM, stream);
  }
  cast_w(cW1, cW1_bf, CELLD * CELLD, stream);
  cast_w(cW2, cW2_bf, CDIM * CELLD, stream);
  cast_w(dW1, dW1_bf, DD * DD, stream);

  // ---- embeddings
  embed_kernel<<<(TOK * DIM) / 256, 256, 0, stream>>>(vidx, pos, deg, vemb, pemb, demb, H);

  // ---- edge MLP -> attention bias
  ln_act_kernel<<<BSZ, 256, 0, stream>>>(dist, e_ln0g, e_ln0b, z_bf, EDGE, 0, 1);
  gemm_bf16_kernel<<<dim3(BSZ / GBM, EDGE / GBN), 256, 0, stream>>>(
      z_bf, eW1_bf, e_b1, nullptr, z_f32, BSZ, EDGE, EDGE, 0, 0);
  ln_act_kernel<<<BSZ, 256, 0, stream>>>(z_f32, e_ln1g, e_ln1b, z_bf, EDGE, 1, 1);
  gemm_bf16_kernel<<<dim3(BSZ / GBM, EDGE / GBN), 256, 0, stream>>>(
      z_bf, eW2_bf, e_b2, nullptr, z_f32, BSZ, EDGE, EDGE, 0, 0);
  ln_act_kernel<<<BSZ, 256, 0, stream>>>(z_f32, e_ln2g, e_ln2b, biasB, EDGE, 2, 0);

  // ---- transformer layers
  dim3 gproj(TOK / GBM, DIM / GBN);
  for (int l = 0; l < 4; ++l) {
    ln_act_kernel<<<TOK, 256, 0, stream>>>(H, L[l].n1g, L[l].n1b, xn_bf, DIM, 0, 1);
    gemm_bf16_kernel<<<gproj, 256, 0, stream>>>(xn_bf, lw_bf[l][0], L[l].bq, nullptr,
                                                q_bf, TOK, DIM, DIM, 0, 1);
    gemm_bf16_kernel<<<gproj, 256, 0, stream>>>(xn_bf, lw_bf[l][1], L[l].bk, nullptr,
                                                k_bf, TOK, DIM, DIM, 0, 1);
    gemm_bf16_kernel<<<gproj, 256, 0, stream>>>(xn_bf, lw_bf[l][2], L[l].bv, nullptr,
                                                v_bf, TOK, DIM, DIM, 0, 1);
    attn_kernel<<<HEADS * BSZ, 128, 0, stream>>>(q_bf, k_bf, v_bf, biasB, mask, at_bf);
    gemm_bf16_kernel<<<gproj, 256, 0, stream>>>(at_bf, lw_bf[l][3], L[l].bo, H,
                                                H, TOK, DIM, DIM, 0, 0);
    ln_act_kernel<<<TOK, 256, 0, stream>>>(H, L[l].n2g, L[l].n2b, xn_bf, DIM, 0, 1);
    gemm_bf16_kernel<<<gproj, 256, 0, stream>>>(xn_bf, lw_bf[l][4], L[l].bf1, nullptr,
                                                f1_bf, TOK, DIM, DIM, 1, 1);
    gemm_bf16_kernel<<<gproj, 256, 0, stream>>>(f1_bf, lw_bf[l][5], L[l].bf2, H,
                                                H, TOK, DIM, DIM, 0, 0);
  }

  // ---- cell encoder
  cell_gather_kernel<<<BSZ, CELLD, 0, stream>>>(cidx, cell_feat, cf_f32);
  cast_bf16_kernel<<<(BSZ * CELLD + 255) / 256, 256, 0, stream>>>(cf_f32, cf_bf, BSZ * CELLD);
  gemm_bf16_kernel<<<dim3(BSZ / GBM, CELLD / GBN), 256, 0, stream>>>(
      cf_bf, cW1_bf, cb1, nullptr, g1_f32, BSZ, CELLD, CELLD, 0, 0);
  ln_act_kernel<<<BSZ, 256, 0, stream>>>(g1_f32, cln1g, cln1b, g1_bf, CELLD, 1, 1);
  gemm_bf16_kernel<<<dim3(BSZ / GBM, CDIM / GBN), 256, 0, stream>>>(
      g1_bf, cW2_bf, cb2, nullptr, g2_f32, BSZ, CDIM, CELLD, 0, 0);
  ln_act_kernel<<<BSZ, 256, 0, stream>>>(g2_f32, cln2g, cln2b, gln, CDIM, 0, 0);

  // ---- decoder
  dec_concat_kernel<<<BSZ, 256, 0, stream>>>(gln, H, lnfg, lnfb, xcat);
  gemm_bf16_kernel<<<dim3(BSZ / GBM, DD / GBN), 256, 0, stream>>>(
      xcat, dW1_bf, db1, nullptr, h1_bf, BSZ, DD, DD, 1, 1);
  dec_final_kernel<<<BSZ, 128, 0, stream>>>(h1_bf, dW2, db2, (float*)d_out);
}
